// MultiHeadAttention_44573170598198
// MI455X (gfx1250) — matrix operand order repaired and checked by reading, not compiled, not run on hardware
//
#include <hip/hip_runtime.h>

typedef __attribute__((ext_vector_type(16))) __bf16 v16bf;
typedef __attribute__((ext_vector_type(8)))  float  v8f;
typedef int v4i_vec __attribute__((vector_size(16)));

#define EMB   768
#define NSEQ  2048
#define BATCH 8
#define HEADS 8
#define HD    96

#if __has_builtin(__builtin_amdgcn_global_load_async_to_lds_b128)
#define HAVE_ASYNC_LDS 1
#else
#define HAVE_ASYNC_LDS 0
#endif

__device__ __forceinline__ void cp16_async(void* lds, const void* g) {
#if HAVE_ASYNC_LDS
  __builtin_amdgcn_global_load_async_to_lds_b128(
      (__attribute__((address_space(1))) v4i_vec*)(void*)g,
      (__attribute__((address_space(3))) v4i_vec*)lds, 0, 0);
#else
  *(int4*)lds = *(const int4*)g;
#endif
}

__device__ __forceinline__ void async_wait0() {
#if HAVE_ASYNC_LDS
#if __has_builtin(__builtin_amdgcn_s_wait_asynccnt)
  __builtin_amdgcn_s_wait_asynccnt(0);
#else
  asm volatile("s_wait_asynccnt 0x0" ::: "memory");
#endif
#endif
}

__device__ __forceinline__ unsigned f2bf(float f) {
  unsigned u = __float_as_uint(f);
  return (u + 0x7FFFu + ((u >> 16) & 1u)) >> 16;   // round-to-nearest-even bf16
}

union FragU { v16bf bf; unsigned u[8]; };

// A fragment (16x32 bf16): lanes 0-15 rows M=0..15 with K pairs {0,1..6,7 | 16..23},
// lanes 16-31 same rows with K pairs {8..15 | 24..31}.  LDS tile is [row][k], k-contiguous.
__device__ __forceinline__ v16bf load_fragA(const unsigned short* lds, int rowBase,
                                            int rowStride, int kOff) {
  const int lane = threadIdx.x & 31;
  const int m    = lane & 15;
  const int kb   = (lane >> 4) << 3;            // 0 or 8
  FragU f;
  const unsigned short* p = lds + (rowBase + m) * rowStride + kOff + kb;
#pragma unroll
  for (int q = 0; q < 8; ++q) {
    const int k = ((q & 4) << 2) + ((q & 3) << 1);   // 0,2,4,6,16,18,20,22
    f.u[q] = *(const unsigned*)(p + k);
  }
  return f.bf;
}

// B fragment (32x16 bf16): lane holds column n=lane&15; lanes 0-15 K={0..7 | 16..23},
// lanes 16-31 K={8..15 | 24..31}.  LDS tile is [col][k].
__device__ __forceinline__ v16bf load_fragB(const unsigned short* lds, int colBase,
                                            int rowStride, int kOff) {
  const int lane = threadIdx.x & 31;
  const int n    = lane & 15;
  const int kb   = (lane >> 4) << 3;            // 0 or 8
  FragU f;
  const unsigned short* p = lds + (colBase + n) * rowStride + kOff + kb;
#pragma unroll
  for (int q = 0; q < 8; ++q)
    f.u[q] = *(const unsigned*)(p + ((q & 4) << 2) + ((q & 3) << 1));
  return f.bf;
}

__device__ __forceinline__ v8f wmma_bf16(v16bf a, v16bf b, v8f c) {
  return __builtin_amdgcn_wmma_f32_16x16x32_bf16(false, a, false, b, (short)0, c,
                                                 false, false);
}

// ---------------------------------------------------------------- conversion
__global__ void __launch_bounds__(256) f32_to_bf16_kernel(const float* __restrict__ in,
                                                          unsigned short* __restrict__ out,
                                                          int n) {
  int i = blockIdx.x * 256 + threadIdx.x;
  const int stride = gridDim.x * 256;
  for (; i < n; i += stride) out[i] = (unsigned short)f2bf(in[i]);
}

// ------------------------------------------------------------------ QKV GEMM
// X[16384x768]bf16 @ W[768x2304]bf16 + bias -> scatter into Q/K/V [B,H,N,D] bf16
__global__ void __launch_bounds__(256) qkv_gemm_kernel(
    const unsigned short* __restrict__ X, const unsigned short* __restrict__ W,
    const float* __restrict__ bias, unsigned short* __restrict__ Qo,
    unsigned short* __restrict__ Ko, unsigned short* __restrict__ Vo) {
  __shared__ alignas(16) unsigned short lA[128 * 32];   // [m][k]
  __shared__ alignas(16) unsigned short lB[128 * 32];   // [n][k]
  const int t = threadIdx.x, lane = t & 31, wave = t >> 5;
  const int wM = wave >> 1, wN = wave & 1;
  const int blockM = blockIdx.y * 128;
  const int blockN = blockIdx.x * 128;
  const v8f zero = {0.f, 0.f, 0.f, 0.f, 0.f, 0.f, 0.f, 0.f};
  v8f acc[2][4];
#pragma unroll
  for (int i = 0; i < 2; ++i)
#pragma unroll
    for (int j = 0; j < 4; ++j) acc[i][j] = zero;

  const unsigned* Wd = (const unsigned*)W;

  for (int kk = 0; kk < 768; kk += 32) {
    __syncthreads();
#pragma unroll
    for (int j = 0; j < 2; ++j) {              // A: 512 x 16B async -> LDS [m][k]
      const int c = t + j * 256;
      const int m = c >> 2, k8 = (c & 3) * 8;
      cp16_async((char*)lA + c * 16, X + (size_t)(blockM + m) * EMB + kk + k8);
    }
#pragma unroll
    for (int j = 0; j < 8; ++j) {              // B: 2048 dwords, transpose to [n][k]
      const int i = t + j * 256;
      const int k = i >> 6, nd = i & 63;
      const unsigned wv = Wd[(kk + k) * 1152 + (blockN >> 1) + nd];
      const int n = nd * 2;
      lB[n * 32 + k]       = (unsigned short)(wv & 0xFFFFu);
      lB[(n + 1) * 32 + k] = (unsigned short)(wv >> 16);
    }
    async_wait0();
    __syncthreads();
    v16bf af[2], bf[4];
#pragma unroll
    for (int mt = 0; mt < 2; ++mt) af[mt] = load_fragA(lA, wM * 32 + mt * 16, 32, 0);
#pragma unroll
    for (int nt = 0; nt < 4; ++nt) bf[nt] = load_fragB(lB, wN * 64 + nt * 16, 32, 0);
#pragma unroll
    for (int mt = 0; mt < 2; ++mt)
#pragma unroll
      for (int nt = 0; nt < 4; ++nt) acc[mt][nt] = wmma_bf16(af[mt], bf[nt], acc[mt][nt]);
  }

  const int hi = (lane >> 4) << 3;
  const int cn = lane & 15;
#pragma unroll
  for (int mt = 0; mt < 2; ++mt)
#pragma unroll
    for (int nt = 0; nt < 4; ++nt) {
      const int col   = blockN + wN * 64 + nt * 16 + cn;
      const float bv  = bias[col];
      const int which = col / EMB;
      const int e     = col % EMB;
      const int h     = e / HD, d = e % HD;
      unsigned short* dst = (which == 0) ? Qo : ((which == 1) ? Ko : Vo);
#pragma unroll
      for (int r = 0; r < 8; ++r) {
        const int row = blockM + wM * 32 + mt * 16 + r + hi;
        const int b = row >> 11, n = row & 2047;
        dst[((b * HEADS + h) * NSEQ + n) * HD + d] = (unsigned short)f2bf(acc[mt][nt][r] + bv);
      }
    }
}

// ---------------------------------------------------------------- attention
// Flash-style; one block = 128 query rows of one (b,h); wave owns 16 rows.
__global__ void __launch_bounds__(256) attention_kernel(
    const unsigned short* __restrict__ Qg, const unsigned short* __restrict__ Kg,
    const unsigned short* __restrict__ Vg, unsigned short* __restrict__ Og) {
  __shared__ alignas(16) unsigned short sQ[128 * 96];   // [m][d]
  __shared__ alignas(16) unsigned short sK[64 * 96];    // [n][d]
  __shared__ alignas(16) unsigned short sV[96 * 64];    // [d][n]  (transposed)
  __shared__ alignas(16) unsigned short sP[8 * 16 * 64];// per-wave P tiles

  const int t = threadIdx.x, lane = t & 31, wave = t >> 5;
  const int hi = (lane >> 4) << 3;
  const int cn = lane & 15;
  const int bh = blockIdx.x >> 4;           // (b*H+h)
  const int qblk = blockIdx.x & 15;
  const unsigned long long base = (unsigned long long)bh * NSEQ * HD;
  const unsigned short* Qp = Qg + base;
  const unsigned short* Kp = Kg + base;
  const unsigned* Vd = (const unsigned*)(Vg + base);

#pragma unroll
  for (int j = 0; j < 6; ++j) {             // Q block: 1536 x 16B async, contiguous
    const int c = t + j * 256;
    cp16_async((char*)sQ + c * 16, Qp + (size_t)qblk * 12288 + c * 8);
  }

  float mS[8], lS[8];
  v8f oacc[6];
  const v8f zero = {0.f, 0.f, 0.f, 0.f, 0.f, 0.f, 0.f, 0.f};
#pragma unroll
  for (int r = 0; r < 8; ++r) { mS[r] = -1e30f; lS[r] = 0.f; }
#pragma unroll
  for (int dt = 0; dt < 6; ++dt) oacc[dt] = zero;
  const float scale = 0.03608439182435161f;   // 1/sqrt(768)

  for (int kb = 0; kb < 32; ++kb) {
    __syncthreads();
#pragma unroll
    for (int j = 0; j < 3; ++j) {           // K tile: 768 x 16B async, contiguous
      const int c = t + j * 256;
      cp16_async((char*)sK + c * 16, Kp + (size_t)kb * 6144 + c * 8);
    }
#pragma unroll
    for (int j = 0; j < 12; ++j) {          // V tile: transpose to [d][n]
      const int i = t + j * 256;
      const int n = i / 48, dd = i % 48;
      const unsigned vv = Vd[kb * 3072 + i];
      const int d = dd * 2;
      sV[d * 64 + n]       = (unsigned short)(vv & 0xFFFFu);
      sV[(d + 1) * 64 + n] = (unsigned short)(vv >> 16);
    }
    async_wait0();
    __syncthreads();

    // S = Q K^T  (16 x 64 per wave)
    v8f s[4];
#pragma unroll
    for (int nt = 0; nt < 4; ++nt) {
      v8f c = zero;
#pragma unroll
      for (int ks = 0; ks < 3; ++ks) {
        v16bf a = load_fragA(sQ, wave * 16, 96, ks * 32);
        v16bf b = load_fragB(sK, nt * 16, 96, ks * 32);
        c = wmma_bf16(a, b, c);
      }
      s[nt] = c;
    }

    // online softmax, write P tile (bf16) to wave-private LDS
#pragma unroll
    for (int r = 0; r < 8; ++r) {
      float sv[4], lm = -1e30f;
#pragma unroll
      for (int nt = 0; nt < 4; ++nt) { sv[nt] = s[nt][r] * scale; lm = fmaxf(lm, sv[nt]); }
#pragma unroll
      for (int off = 8; off >= 1; off >>= 1) lm = fmaxf(lm, __shfl_xor(lm, off, 32));
      const float mNew = fmaxf(mS[r], lm);
      const float corr = __expf(mS[r] - mNew);
      float rs = 0.f;
      const int rowLds = r + hi;
#pragma unroll
      for (int nt = 0; nt < 4; ++nt) {
        const float p = __expf(sv[nt] - mNew);
        rs += p;
        sP[(wave * 16 + rowLds) * 64 + nt * 16 + cn] = (unsigned short)f2bf(p);
      }
#pragma unroll
      for (int off = 8; off >= 1; off >>= 1) rs += __shfl_xor(rs, off, 32);
      lS[r] = lS[r] * corr + rs;
      mS[r] = mNew;
#pragma unroll
      for (int dt = 0; dt < 6; ++dt) oacc[dt][r] *= corr;
    }
    asm volatile("s_wait_dscnt 0x0" ::: "memory");   // wave-private P RAW in LDS

    // O += P V   (A = P[16x64], B = V^T tiles)
    v16bf pa0 = load_fragA(sP + wave * 16 * 64, 0, 64, 0);
    v16bf pa1 = load_fragA(sP + wave * 16 * 64, 0, 64, 32);
#pragma unroll
    for (int dt = 0; dt < 6; ++dt) {
      oacc[dt] = wmma_bf16(pa0, load_fragB(sV, dt * 16, 64, 0),  oacc[dt]);
      oacc[dt] = wmma_bf16(pa1, load_fragB(sV, dt * 16, 64, 32), oacc[dt]);
    }
  }

  // normalize and write attn_out [B*N][E] bf16, e = h*96 + d
  const int b = bh >> 3, h = bh & 7;
#pragma unroll
  for (int dt = 0; dt < 6; ++dt)
#pragma unroll
    for (int r = 0; r < 8; ++r) {
      const int qrow = qblk * 128 + wave * 16 + r + hi;
      const float v = oacc[dt][r] / lS[r];
      Og[(unsigned long long)(b * NSEQ + qrow) * EMB + h * HD + dt * 16 + cn] =
          (unsigned short)f2bf(v);
    }
}

// ------------------------------------------------------------------ proj GEMM
// A[16384x768]bf16 @ W[768x768]bf16 + bias -> f32 out
__global__ void __launch_bounds__(256) proj_gemm_kernel(
    const unsigned short* __restrict__ A, const unsigned short* __restrict__ W,
    const float* __restrict__ bias, float* __restrict__ out) {
  __shared__ alignas(16) unsigned short lA[128 * 32];
  __shared__ alignas(16) unsigned short lB[128 * 32];
  const int t = threadIdx.x, lane = t & 31, wave = t >> 5;
  const int wM = wave >> 1, wN = wave & 1;
  const int blockM = blockIdx.y * 128;
  const int blockN = blockIdx.x * 128;
  const v8f zero = {0.f, 0.f, 0.f, 0.f, 0.f, 0.f, 0.f, 0.f};
  v8f acc[2][4];
#pragma unroll
  for (int i = 0; i < 2; ++i)
#pragma unroll
    for (int j = 0; j < 4; ++j) acc[i][j] = zero;

  const unsigned* Wd = (const unsigned*)W;
  for (int kk = 0; kk < 768; kk += 32) {
    __syncthreads();
#pragma unroll
    for (int j = 0; j < 2; ++j) {              // A: async 16B copies
      const int c = t + j * 256;
      const int m = c >> 2, k8 = (c & 3) * 8;
      cp16_async((char*)lA + c * 16, A + (size_t)(blockM + m) * EMB + kk + k8);
    }
#pragma unroll
    for (int j = 0; j < 8; ++j) {
      const int i = t + j * 256;
      const int k = i >> 6, nd = i & 63;
      const unsigned wv = Wd[(kk + k) * 384 + (blockN >> 1) + nd];
      const int n = nd * 2;
      lB[n * 32 + k]       = (unsigned short)(wv & 0xFFFFu);
      lB[(n + 1) * 32 + k] = (unsigned short)(wv >> 16);
    }
    async_wait0();
    __syncthreads();
    v16bf af[2], bf[4];
#pragma unroll
    for (int mt = 0; mt < 2; ++mt) af[mt] = load_fragA(lA, wM * 32 + mt * 16, 32, 0);
#pragma unroll
    for (int nt = 0; nt < 4; ++nt) bf[nt] = load_fragB(lB, wN * 64 + nt * 16, 32, 0);
#pragma unroll
    for (int mt = 0; mt < 2; ++mt)
#pragma unroll
      for (int nt = 0; nt < 4; ++nt) acc[mt][nt] = wmma_bf16(af[mt], bf[nt], acc[mt][nt]);
  }

  const int hi = (lane >> 4) << 3;
  const int cn = lane & 15;
#pragma unroll
  for (int mt = 0; mt < 2; ++mt)
#pragma unroll
    for (int nt = 0; nt < 4; ++nt) {
      const int col = blockN + wN * 64 + nt * 16 + cn;
      const float bv = bias[col];
#pragma unroll
      for (int r = 0; r < 8; ++r) {
        const int row = blockM + wM * 32 + mt * 16 + r + hi;
        out[row * EMB + col] = acc[mt][nt][r] + bv;
      }
    }
}

// ------------------------------------------------------------------- launch
extern "C" void kernel_launch(void* const* d_in, const int* in_sizes, int n_in,
                              void* d_out, int out_size, void* d_ws, size_t ws_size,
                              hipStream_t stream) {
  const float* x     = (const float*)d_in[0];
  const float* Wqkv  = (const float*)d_in[1];
  const float* bqkv  = (const float*)d_in[2];
  const float* Wproj = (const float*)d_in[3];
  const float* bproj = (const float*)d_in[4];
  float* out = (float*)d_out;

  const size_t N_X  = (size_t)BATCH * NSEQ * EMB;   // 12,582,912
  const size_t N_WQ = (size_t)EMB * 3 * EMB;        //  1,769,472
  const size_t N_WP = (size_t)EMB * EMB;            //    589,824
  const size_t N_QKV = (size_t)BATCH * HEADS * NSEQ * HD;  // 12,582,912 each

  char* ws = (char*)d_ws;
  unsigned short* xb    = (unsigned short*)ws;                       // also reused as attn_out
  unsigned short* wqkvb = (unsigned short*)(ws + 2 * N_X);
  unsigned short* wprojb= (unsigned short*)(ws + 2 * (N_X + N_WQ));
  unsigned short* qb    = (unsigned short*)(ws + 2 * (N_X + N_WQ + N_WP));
  unsigned short* kb    = qb + N_QKV;
  unsigned short* vb    = kb + N_QKV;
  unsigned short* attnb = xb;   // x consumed by QKV GEMM before attention writes

  f32_to_bf16_kernel<<<2048, 256, 0, stream>>>(x, xb, (int)N_X);
  f32_to_bf16_kernel<<<1024, 256, 0, stream>>>(Wqkv, wqkvb, (int)N_WQ);
  f32_to_bf16_kernel<<<512, 256, 0, stream>>>(Wproj, wprojb, (int)N_WP);

  qkv_gemm_kernel<<<dim3(18, 128), 256, 0, stream>>>(xb, wqkvb, bqkv, qb, kb, vb);
  attention_kernel<<<1024, 256, 0, stream>>>(qb, kb, vb, attnb);
  proj_gemm_kernel<<<dim3(6, 128), 256, 0, stream>>>(attnb, wprojb, bproj, out);
}
